// CRFLoss_12275016532172
// MI455X (gfx1250) — matrix operand order repaired and checked by reading, not compiled, not run on hardware
//
#include <hip/hip_runtime.h>

typedef __attribute__((ext_vector_type(16))) _Float16 v16h;
typedef __attribute__((ext_vector_type(8)))  float    v8f;

#define B_TOT 512
#define T_LEN 512
#define C_DIM 64
#define BTILE 16          // batch rows per block (one WMMA M-tile)
#define NWAVES 4          // 4 waves: one 16-column N-tile each
#define TPB (NWAVES * 32)

// ---------------------------------------------------------------------------
// out[0] = 0  (single scalar accumulator; kernels below atomically add)
// ---------------------------------------------------------------------------
__global__ void crf_zero(float* out) { out[0] = 0.0f; }

// ---------------------------------------------------------------------------
// Gold-path score: one thread per batch row. score_b accumulated, then
// atomicAdd(out, -score_b)  (output = sum_b (log_z_b - score_b)).
// ---------------------------------------------------------------------------
__global__ __launch_bounds__(256) void crf_score(
    const float* __restrict__ emis, const int* __restrict__ tags,
    const unsigned char* __restrict__ mask,
    const float* __restrict__ startT, const float* __restrict__ endT,
    const float* __restrict__ trans, float* __restrict__ out)
{
    int b = blockIdx.x * blockDim.x + threadIdx.x;
    if (b >= B_TOT) return;
    const int TC = T_LEN * C_DIM;

    int   tprev = tags[b * T_LEN];
    float sc    = startT[tprev] + emis[b * TC + tprev];
    int   cnt   = mask[b * T_LEN] ? 1 : 0;

    for (int t = 1; t < T_LEN; ++t) {
        int tg = tags[b * T_LEN + t];
        if (mask[b * T_LEN + t]) {
            sc += emis[b * TC + t * C_DIM + tg] + trans[tprev * C_DIM + tg];
            ++cnt;
        }
        tprev = tg;
    }
    int last = tags[b * T_LEN + (cnt - 1)];   // tags[b, seq_end]
    sc += endT[last];
    atomicAdd(out, -sc);
}

// ---------------------------------------------------------------------------
// Forward algorithm (log partition) via WMMA in the exp-domain:
//   alpha_next[b,j] = e[b,t,j] + m[b] + log( sum_i exp(alpha[b,i]-m[b]) * E[i,j] )
// with E = exp(transitions) held as constant f16 B-fragments in registers.
// One block = 16 batch rows; wave w owns output columns [16w, 16w+16).
// ---------------------------------------------------------------------------
__global__ __launch_bounds__(TPB) void crf_forward(
    const float* __restrict__ emis, const unsigned char* __restrict__ mask,
    const float* __restrict__ startT, const float* __restrict__ endT,
    const float* __restrict__ trans, float* __restrict__ out)
{
    __shared__ _Float16 Eh[C_DIM][C_DIM];        // exp(transitions), f16
    __shared__ float    alpha[BTILE][C_DIM + 4]; // padded f32 alpha
    __shared__ _Float16 Ah[BTILE][C_DIM];        // exp(alpha - m), f16, row-major
    __shared__ float    mrow[BTILE];             // per-row max

    const int tid    = threadIdx.x;
    const int lane   = tid & 31;
    const int wave   = tid >> 5;
    const int laneN  = lane & 15;   // col within tile / A-matrix row M
    const int laneHi = lane >> 4;   // half-wave selector
    const int n0     = wave * 16;   // this wave's first output column
    const int bBase  = blockIdx.x * BTILE;
    const int TC     = T_LEN * C_DIM;

    // --- E = exp(transitions) into LDS (B-matrix source, constant over T) ---
    for (int k = tid; k < C_DIM * C_DIM; k += TPB)
        Eh[k >> 6][k & 63] = (_Float16)__expf(trans[k]);

    // --- alpha_0 = start + emissions[:,0,:] ---
    for (int k = tid; k < BTILE * C_DIM; k += TPB) {
        int r = k >> 6, j = k & 63;
        alpha[r][j] = startT[j] + emis[(bBase + r) * TC + j];
    }
    __syncthreads();

    // --- per-wave B fragments (B KxN, contiguous k order shared with A: half h <-> K = 16*laneHi + h,
    //     lane%16 <-> N), hoisted out of the T loop ---
    v16h bf0, bf1;
    #pragma unroll
    for (int h = 0; h < 16; ++h) {
        bf0[h] = Eh[     laneHi * 16 + h][n0 + laneN];   // K =  0..31
        bf1[h] = Eh[32 + laneHi * 16 + h][n0 + laneN];   // K = 32..63
    }

    for (int t = 1; t < T_LEN; ++t) {
        __syncthreads();                        // prior alpha writes visible
        if (tid < BTILE) {                      // row max
            float mx = alpha[tid][0];
            #pragma unroll
            for (int j = 1; j < C_DIM; ++j) mx = fmaxf(mx, alpha[tid][j]);
            mrow[tid] = mx;
        }
        __syncthreads();
        for (int k = tid; k < BTILE * C_DIM; k += TPB) {   // A staging
            int r = k >> 6, j = k & 63;
            Ah[r][j] = (_Float16)__expf(alpha[r][j] - mrow[r]);
        }
        __syncthreads();

        // A fragments (16x32, contiguous k order, same as B): row M = lane%16;
        // halves 0..15 = K[base..base+15],
        // base = 16*laneHi (+32 for the second K-chunk). Contiguous 16B chunks.
        const _Float16* arow = &Ah[laneN][0];
        const int base0 = laneHi * 16;
        v16h a0, a1;
        #pragma unroll
        for (int h = 0; h < 8; ++h) {
            a0[h]     = arow[base0 + h];
            a0[h + 8] = arow[base0 + 8 + h];
            a1[h]     = arow[32 + base0 + h];
            a1[h + 8] = arow[32 + base0 + 8 + h];
        }

        v8f acc = {};
        acc = __builtin_amdgcn_wmma_f32_16x16x32_f16(false, a0, false, bf0,
                                                     (short)0, acc, false, false);
        acc = __builtin_amdgcn_wmma_f32_16x16x32_f16(false, a1, false, bf1,
                                                     (short)0, acc, false, false);

        // hint next timestep's emissions into cache while WMMA drains
        if (t + 1 < T_LEN)
            __builtin_prefetch(&emis[(bBase + laneHi * 8 + (laneN & 7)) * TC +
                                     (t + 1) * C_DIM], 0, 3);

        // alpha update (f32 C/D layout: VGPR r <-> row r + 8*laneHi, col = n0+lane%16)
        const int j = n0 + laneN;
        #pragma unroll
        for (int r = 0; r < 8; ++r) {
            int   br   = r + laneHi * 8;
            float cand = emis[(bBase + br) * TC + t * C_DIM + j] +
                         mrow[br] + __logf(acc[r]);
            float oldv = alpha[br][j];
            alpha[br][j] = mask[(bBase + br) * T_LEN + t] ? cand : oldv;
        }
    }

    __syncthreads();
    // log_z[b] = logsumexp_j(alpha[b,j] + end[j]); accumulate +log_z into out
    if (tid < BTILE) {
        int   br = tid;
        float mx = alpha[br][0] + endT[0];
        #pragma unroll
        for (int j = 1; j < C_DIM; ++j) mx = fmaxf(mx, alpha[br][j] + endT[j]);
        float s = 0.0f;
        #pragma unroll
        for (int j = 0; j < C_DIM; ++j) s += __expf(alpha[br][j] + endT[j] - mx);
        atomicAdd(out, mx + __logf(s));
    }
}

// ---------------------------------------------------------------------------
extern "C" void kernel_launch(void* const* d_in, const int* in_sizes, int n_in,
                              void* d_out, int out_size, void* d_ws, size_t ws_size,
                              hipStream_t stream)
{
    const float*         emis   = (const float*)d_in[0];
    const int*           tags   = (const int*)d_in[1];
    const unsigned char* mask   = (const unsigned char*)d_in[2];
    const float*         startT = (const float*)d_in[3];
    const float*         endT   = (const float*)d_in[4];
    const float*         trans  = (const float*)d_in[5];
    float*               out    = (float*)d_out;

    crf_zero<<<1, 1, 0, stream>>>(out);
    crf_score<<<(B_TOT + 255) / 256, 256, 0, stream>>>(emis, tags, mask,
                                                       startT, endT, trans, out);
    crf_forward<<<B_TOT / BTILE, TPB, 0, stream>>>(emis, mask,
                                                   startT, endT, trans, out);
}
